// MultiHeadDuelingDQN_26560077758995
// MI455X (gfx1250) — matrix operand order repaired and checked by reading, not compiled, not run on hardware
//
#include <hip/hip_runtime.h>

// ---------------------------------------------------------------------------
// MultiHeadDuelingDQN forward for MI455X (gfx1250).
// bf16 WMMA (f32 accum) for all GEMMs; the dominant adv_rec term is
// algebraically collapsed (head-sum commutes with the GEMM; the dueling mean
// of rec_global is exactly zero) into one streaming NT reduction over Wru
// (524 MB -> HBM bound, ~22.5us @ 23.3 TB/s) plus a [512,4000] GEMM.
// GEMM1 (K=26240) is split-K (grid.z=10, 320 blocks).  Double-buffered LDS,
// one barrier per k-step, next-tile global loads overlap WMMA.
// ---------------------------------------------------------------------------

#define BATCH   256
#define HID     512
#define NHEAD   64
#define RDIM    4000
#define MDIM    10
#define STATE_D 26240
#define OUTSTR  8640          // 2*RDIM + NHEAD*MDIM
#define KSPLIT  10
#define KCHUNK  2624          // STATE_D / KSPLIT, multiple of 32

typedef __attribute__((ext_vector_type(16))) __bf16 v16bf;
typedef __attribute__((ext_vector_type(2)))  __bf16 v2bf;
typedef __attribute__((ext_vector_type(8)))  float  v8f;
typedef __attribute__((ext_vector_type(4)))  float  v4f;
typedef __attribute__((ext_vector_type(2)))  float  v2f;

union BF16Frag { v16bf v; unsigned int u[8]; };

// Two floats -> packed bf16 pair (RNE); pattern-matches v_cvt_pk_bf16_f32.
__device__ __forceinline__ unsigned int pack2bf(float x, float y) {
    v2f f; f[0] = x; f[1] = y;
    union { v2bf v; unsigned int u; } c;
    c.v = __builtin_convertvector(f, v2bf);
    return c.u;
}

// ---------------------------------------------------------------------------
// Streaming reduction: Wsum[h][r] = sum_n Wru[n][h][r].  524 MB, read once,
// fully coalesced, NON-TEMPORAL (zero reuse; keep L2 for the GEMM operands).
// ---------------------------------------------------------------------------
__global__ __launch_bounds__(256) void reduce_wru(const v4f* __restrict__ Wru,
                                                  v4f* __restrict__ Wsum) {
    const int idx = blockIdx.x * 256 + threadIdx.x;           // 0..511999
    const v4f* p = Wru + idx;
    v4f s0 = {0.f, 0.f, 0.f, 0.f};
    v4f s1 = {0.f, 0.f, 0.f, 0.f};
#pragma unroll 8
    for (int n = 0; n < NHEAD; n += 2) {
        s0 += __builtin_nontemporal_load(p);
        s1 += __builtin_nontemporal_load(p + (HID * RDIM / 4));
        p += 2 * (HID * RDIM / 4);
    }
    Wsum[idx] = s0 + s1;
}

// c[h] = sum_r Wsum[h][r]
__global__ __launch_bounds__(256) void reduce_c(const float* __restrict__ Wsum,
                                                float* __restrict__ c) {
    __shared__ float red[256];
    const int h = blockIdx.x, tid = threadIdx.x;
    float s = 0.f;
    for (int r = tid; r < RDIM; r += 256) s += Wsum[(size_t)h * RDIM + r];
    red[tid] = s; __syncthreads();
    for (int off = 128; off > 0; off >>= 1) {
        if (tid < off) red[tid] += red[tid + off];
        __syncthreads();
    }
    if (tid == 0) c[h] = red[0];
}

// bsum[r] = sum_n bru[n][r]
__global__ __launch_bounds__(256) void reduce_bsum(const float* __restrict__ bru,
                                                   float* __restrict__ bsum) {
    const int r = blockIdx.x * 256 + threadIdx.x;
    if (r >= RDIM) return;
    float s = 0.f;
#pragma unroll 8
    for (int n = 0; n < NHEAD; ++n) s += bru[n * RDIM + r];
    bsum[r] = s;
}

// tot = sum_r bsum[r]
__global__ __launch_bounds__(256) void sum_total(const float* __restrict__ bsum,
                                                 float* __restrict__ tot) {
    __shared__ float red[256];
    const int tid = threadIdx.x;
    float s = 0.f;
    for (int r = tid; r < RDIM; r += 256) s += bsum[r];
    red[tid] = s; __syncthreads();
    for (int off = 128; off > 0; off >>= 1) {
        if (tid < off) red[tid] += red[tid + off];
        __syncthreads();
    }
    if (tid == 0) tot[0] = red[0];
}

// Bvut[h][n] = Wvu[n][h]
__global__ __launch_bounds__(256) void pack_vu(const float* __restrict__ Wvu,
                                               float* __restrict__ Bt) {
    const int o = blockIdx.x * 256 + threadIdx.x;
    if (o >= HID * NHEAD) return;
    const int h = o >> 6, n = o & 63;
    Bt[o] = Wvu[n * HID + h];
}

// Bau[h][n*10+m] = Wau[n][h][m]
__global__ __launch_bounds__(256) void pack_au(const float* __restrict__ Wau,
                                               float* __restrict__ Bp) {
    const int o = blockIdx.x * 256 + threadIdx.x;
    if (o >= HID * NHEAD * MDIM) return;
    const int h = o / (NHEAD * MDIM), col = o % (NHEAD * MDIM);
    const int n = col / MDIM, m = col % MDIM;
    Bp[o] = Wau[n * (HID * MDIM) + h * MDIM + m];
}

// h1[i] = relu(sum_c part[c][i] + b1[i%HID])   (split-K combine)
__global__ __launch_bounds__(256) void reduce_splitk(const float* __restrict__ part,
                                                     const float* __restrict__ b1,
                                                     float* __restrict__ h1) {
    const int i = blockIdx.x * 256 + threadIdx.x;             // < BATCH*HID
    float s = b1[i & (HID - 1)];
#pragma unroll
    for (int c = 0; c < KSPLIT; ++c) s += part[(size_t)c * (BATCH * HID) + i];
    h1[i] = s > 0.f ? s : 0.f;
}

// ---------------------------------------------------------------------------
// bf16 WMMA GEMM: C[M,N] = op(A[M,K]*B[K,N] (+bias)), f32 accumulate.
// 64x64 block tile, BK=32, 256 threads (8 waves); wave w -> 16-row group w/2,
// 32-col group w%2 -> two v_wmma_f32_16x16x32_bf16 per k-step.
// Double-buffered LDS (one barrier per k-step; next-tile global loads issued
// before the WMMAs).  A staged row-major bf16 (stride 20 uints: 16B-aligned
// b128 fragment reads, bank-conflict-free); B staged transposed as packed
// k-pair uints (exactly the B fragment granularity).  OOB N columns handled
// by pre-zeroing the B buffers once (uniform per-thread store predicate; B
// row pointer advances as a plain induction variable).
// grid.z = split-K chunk.  M = grid.y*64; Kchunk % 32 == 0; N % 16 == 0.
// ---------------------------------------------------------------------------
template <bool RELU>
__global__ __launch_bounds__(256) void gemm_bf16(const float* __restrict__ A, int lda,
                                                 const float* __restrict__ B, int ldb,
                                                 const float* __restrict__ bias,
                                                 float* __restrict__ C, int ldc,
                                                 int N, int Kchunk, int chunkStrideC) {
    __shared__ unsigned int AsU[2][64 * 20];   // [buf][row*20 + kpair]
    __shared__ unsigned int BtU[2][64 * 20];   // [buf][col*20 + kpair]

    const int t     = threadIdx.x;
    const int mBase = blockIdx.y * 64;
    const int nBase = blockIdx.x * 64;
    const int kOff  = blockIdx.z * Kchunk;
    C += (size_t)blockIdx.z * chunkStrideC;

    const int w    = t >> 5, lane = t & 31;
    const int rg   = w >> 1, cg = w & 1;
    const int m0   = rg * 16;
    const int lmod = lane & 15, half = lane >> 4;

    const int atr = t >> 2, ac8 = (t & 3) * 8;   // A tile: 64 rows x 32 k
    const int bkp = t >> 4, bnc = (t & 15) * 4;  // B tile: 16 k-pairs x 64 n

    // zero B buffers once (covers OOB columns which are never written)
    for (int i = t; i < 2 * 64 * 20; i += 256) ((unsigned int*)BtU)[i] = 0u;
    __syncthreads();

    const float* ap = A + (size_t)(mBase + atr) * lda + kOff + ac8;
    const bool  inB = (nBase + bnc) < N;       // N % 4 == 0 for all callers
    const float* bp = B + (size_t)(kOff + 2 * bkp) * ldb + nBase + bnc;

    const int asIdx = atr * 20 + (ac8 >> 1);
    const int btIdx = bnc * 20 + bkp;

    // ---- prefetch tile 0 into buffer 0 ----
    {
        float4 a0 = *(const float4*)ap;
        float4 a1 = *(const float4*)(ap + 4);
        ap += 32;
        unsigned int* asw = &AsU[0][asIdx];
        asw[0] = pack2bf(a0.x, a0.y);
        asw[1] = pack2bf(a0.z, a0.w);
        asw[2] = pack2bf(a1.x, a1.y);
        asw[3] = pack2bf(a1.z, a1.w);
        if (inB) {
            float4 b0 = *(const float4*)bp;
            float4 b1 = *(const float4*)(bp + ldb);
            BtU[0][btIdx +  0] = pack2bf(b0.x, b1.x);
            BtU[0][btIdx + 20] = pack2bf(b0.y, b1.y);
            BtU[0][btIdx + 40] = pack2bf(b0.z, b1.z);
            BtU[0][btIdx + 60] = pack2bf(b0.w, b1.w);
        }
        bp += (size_t)32 * ldb;
    }

    v8f acc0 = {}; v8f acc1 = {};
    const int nk = Kchunk >> 5;

    for (int k = 0; k < nk; ++k) {
        __syncthreads();                       // buf[k&1] ready for all waves
        const int cur = k & 1;
        const bool more = (k + 1) < nk;

        // ---- issue next tile's global loads before the WMMAs ----
        float4 a0 = make_float4(0.f, 0.f, 0.f, 0.f), a1 = a0, b0 = a0, b1 = a0;
        if (more) {
            a0 = *(const float4*)ap;
            a1 = *(const float4*)(ap + 4);
            ap += 32;
            if (inB) {
                b0 = *(const float4*)bp;
                b1 = *(const float4*)(bp + ldb);
            }
            bp += (size_t)32 * ldb;
        }

        // ---- fragments: contiguous k order (k = 16*half + s), same for A and B ----
        // A 16x32: lanes 0-15 K 0..15; lanes 16-31 K 16..31
        BF16Frag af;
        const unsigned int* arow = &AsU[cur][(m0 + lmod) * 20];
#pragma unroll
        for (int i = 0; i < 4; ++i) af.u[i]     = arow[half * 8 + i];
#pragma unroll
        for (int i = 0; i < 4; ++i) af.u[4 + i] = arow[half * 8 + 4 + i];

        // B 32x16: lanes 0-15 K 0..15; lanes 16-31 K 16..31; n = lane%16
        BF16Frag bf0, bf1;
        const unsigned int* br0 = &BtU[cur][(cg * 32 + lmod) * 20 + half * 8];
        const unsigned int* br1 = &BtU[cur][(cg * 32 + 16 + lmod) * 20 + half * 8];
#pragma unroll
        for (int i = 0; i < 8; ++i) bf0.u[i] = br0[i];
#pragma unroll
        for (int i = 0; i < 8; ++i) bf1.u[i] = br1[i];

        acc0 = __builtin_amdgcn_wmma_f32_16x16x32_bf16(false, af.v, false, bf0.v,
                                                       (short)0, acc0, false, false);
        acc1 = __builtin_amdgcn_wmma_f32_16x16x32_bf16(false, af.v, false, bf1.v,
                                                       (short)0, acc1, false, false);

        // ---- stage next tile into the other buffer ----
        if (more) {
            const int nxt = cur ^ 1;
            unsigned int* asw = &AsU[nxt][asIdx];
            asw[0] = pack2bf(a0.x, a0.y);
            asw[1] = pack2bf(a0.z, a0.w);
            asw[2] = pack2bf(a1.x, a1.y);
            asw[3] = pack2bf(a1.z, a1.w);
            if (inB) {
                BtU[nxt][btIdx +  0] = pack2bf(b0.x, b1.x);
                BtU[nxt][btIdx + 20] = pack2bf(b0.y, b1.y);
                BtU[nxt][btIdx + 40] = pack2bf(b0.z, b1.z);
                BtU[nxt][btIdx + 60] = pack2bf(b0.w, b1.w);
            }
        }
    }

    // ---- store: C 16x16 layout: lanes 0-15 M=r, lanes 16-31 M=8+r ----
    const int crow = mBase + m0 + half * 8;
    int ncol = nBase + cg * 32 + lmod;
    if (ncol < N) {
        const float bv = bias ? bias[ncol] : 0.f;
#pragma unroll
        for (int r = 0; r < 8; ++r) {
            float v = acc0[r] + bv;
            if (RELU) v = v > 0.f ? v : 0.f;
            C[(size_t)(crow + r) * ldc + ncol] = v;
        }
    }
    ncol += 16;
    if (ncol < N) {
        const float bv = bias ? bias[ncol] : 0.f;
#pragma unroll
        for (int r = 0; r < 8; ++r) {
            float v = acc1[r] + bv;
            if (RELU) v = v > 0.f ? v : 0.f;
            C[(size_t)(crow + r) * ldc + ncol] = v;
        }
    }
}

// ---------------------------------------------------------------------------
// Finalize kernels (in-place dueling mean subtraction on d_out slices).
// ---------------------------------------------------------------------------
__global__ __launch_bounds__(256) void finalize_cache(float* __restrict__ out,
                                                      const float* __restrict__ h,
                                                      const float* __restrict__ Wvc,
                                                      const float* __restrict__ bvc) {
    __shared__ float red[256];
    __shared__ float sh[2];
    const int b = blockIdx.x, tid = threadIdx.x;
    float* row = out + (size_t)b * OUTSTR;            // q_cache slice
    float s1 = 0.f;
    for (int r = tid; r < RDIM; r += 256) s1 += row[r];
    red[tid] = s1; __syncthreads();
    for (int off = 128; off > 0; off >>= 1) { if (tid < off) red[tid] += red[tid + off]; __syncthreads(); }
    if (tid == 0) sh[0] = red[0] * (1.0f / RDIM);
    __syncthreads();
    float s2 = 0.f;
    for (int k = tid; k < HID; k += 256) s2 += h[(size_t)b * HID + k] * Wvc[k];
    red[tid] = s2; __syncthreads();
    for (int off = 128; off > 0; off >>= 1) { if (tid < off) red[tid] += red[tid + off]; __syncthreads(); }
    if (tid == 0) sh[1] = red[0] + bvc[0];
    __syncthreads();
    const float mean = sh[0], vc = sh[1];
    for (int r = tid; r < RDIM; r += 256) row[r] = vc + row[r] - mean;
}

__global__ __launch_bounds__(256) void finalize_rec(float* __restrict__ out,
                                                    const float* __restrict__ h,
                                                    const float* __restrict__ c,
                                                    const float* __restrict__ tot) {
    __shared__ float red[256];
    __shared__ float sh;
    const int b = blockIdx.x, tid = threadIdx.x;
    float s2 = 0.f;
    for (int k = tid; k < HID; k += 256) s2 += h[(size_t)b * HID + k] * c[k];
    red[tid] = s2; __syncthreads();
    for (int off = 128; off > 0; off >>= 1) { if (tid < off) red[tid] += red[tid + off]; __syncthreads(); }
    if (tid == 0) sh = (red[0] + tot[0]) * (1.0f / RDIM);
    __syncthreads();
    const float s = sh;
    float* row = out + (size_t)b * OUTSTR + RDIM;     // q_rec slice
    for (int r = tid; r < RDIM; r += 256) row[r] -= s;
}

__global__ __launch_bounds__(64) void finalize_assoc(float* __restrict__ out,
                                                     const float* __restrict__ valn) {
    const int b = blockIdx.x, n = threadIdx.x;        // 64 threads = 64 heads
    float* p = out + (size_t)b * OUTSTR + 2 * RDIM + n * MDIM;
    float v[MDIM], s = 0.f;
#pragma unroll
    for (int m = 0; m < MDIM; ++m) { v[m] = p[m]; s += v[m]; }
    const float mean = s * (1.0f / MDIM);
    const float vn = valn[b * NHEAD + n];
#pragma unroll
    for (int m = 0; m < MDIM; ++m) p[m] = vn + v[m] - mean;
}

// ---------------------------------------------------------------------------
// Workspace layout (floats, all float4-aligned)
// ---------------------------------------------------------------------------
static const size_t WSUM_OFF = 0;                      // 512*4000   = 2,048,000
static const size_t H1P_OFF  = 2048000;                // 10*256*512 = 1,310,720
static const size_t H1_OFF   = 3358720;                // 256*512
static const size_t H_OFF    = 3489792;                // 256*512
static const size_t VALN_OFF = 3620864;                // 256*64
static const size_t C_OFF    = 3637248;                // 512
static const size_t BSUM_OFF = 3637760;                // 4000
static const size_t BTOT_OFF = 3641760;                // 1 (padded to 16)
static const size_t BVUT_OFF = 3641776;                // 512*64
static const size_t BAU_OFF  = 3674544;                // 512*640
// total = 4,002,224 floats ~= 16.0 MB

extern "C" void kernel_launch(void* const* d_in, const int* in_sizes, int n_in,
                              void* d_out, int out_size, void* d_ws, size_t ws_size,
                              hipStream_t stream) {
    (void)in_sizes; (void)n_in; (void)out_size; (void)ws_size;
    const float* x   = (const float*)d_in[0];
    const float* W1  = (const float*)d_in[1];
    const float* b1  = (const float*)d_in[2];
    const float* W2  = (const float*)d_in[3];
    const float* b2  = (const float*)d_in[4];
    const float* Wvc = (const float*)d_in[5];
    const float* bvc = (const float*)d_in[6];
    const float* Wac = (const float*)d_in[7];
    const float* bac = (const float*)d_in[8];
    const float* Wvu = (const float*)d_in[9];
    const float* bvu = (const float*)d_in[10];
    const float* Wau = (const float*)d_in[11];
    const float* bau = (const float*)d_in[12];   // [64,10] flat == column order
    const float* Wru = (const float*)d_in[13];
    const float* bru = (const float*)d_in[14];
    float* out = (float*)d_out;
    float* ws  = (float*)d_ws;

    float* Wsum  = ws + WSUM_OFF;
    float* h1par = ws + H1P_OFF;
    float* h1    = ws + H1_OFF;
    float* hbuf  = ws + H_OFF;
    float* valn  = ws + VALN_OFF;
    float* cvec  = ws + C_OFF;
    float* bsum  = ws + BSUM_OFF;
    float* btot  = ws + BTOT_OFF;
    float* Bvut  = ws + BVUT_OFF;
    float* Bau   = ws + BAU_OFF;

    // Precompute (dominant: 524 MB non-temporal Wru stream, HBM bound)
    reduce_wru <<<2000, 256, 0, stream>>>((const v4f*)Wru, (v4f*)Wsum);
    reduce_c   <<<512,  256, 0, stream>>>(Wsum, cvec);
    reduce_bsum<<<16,   256, 0, stream>>>(bru, bsum);
    sum_total  <<<1,    256, 0, stream>>>(bsum, btot);
    pack_vu    <<<128,  256, 0, stream>>>(Wvu, Bvut);
    pack_au    <<<1280, 256, 0, stream>>>(Wau, Bau);

    // Trunk MLP (bf16 WMMA).  GEMM1 split-K over grid.z=10 -> 320 blocks.
    gemm_bf16<false><<<dim3(8, 4, KSPLIT), 256, 0, stream>>>(
        x, STATE_D, W1, HID, nullptr, h1par, HID, HID, KCHUNK, BATCH * HID);
    reduce_splitk<<<BATCH * HID / 256, 256, 0, stream>>>(h1par, b1, h1);
    gemm_bf16<true><<<dim3(8, 4, 1), 256, 0, stream>>>(
        h1, HID, W2, HID, b2, hbuf, HID, HID, HID, 0);

    // Heads -> write straight into d_out slices
    gemm_bf16<false><<<dim3(63, 4, 1), 256, 0, stream>>>(
        hbuf, HID, Wac,  RDIM, bac,  out,            OUTSTR, RDIM, HID, 0);
    gemm_bf16<false><<<dim3(63, 4, 1), 256, 0, stream>>>(
        hbuf, HID, Wsum, RDIM, bsum, out + RDIM,     OUTSTR, RDIM, HID, 0);
    gemm_bf16<false><<<dim3(1, 4, 1),  256, 0, stream>>>(
        hbuf, HID, Bvut, NHEAD, bvu, valn,           NHEAD,  NHEAD, HID, 0);
    gemm_bf16<false><<<dim3(10, 4, 1), 256, 0, stream>>>(
        hbuf, HID, Bau,  NHEAD * MDIM, bau, out + 2 * RDIM, OUTSTR, NHEAD * MDIM, HID, 0);

    // Dueling mean subtractions
    finalize_cache<<<BATCH, 256, 0, stream>>>(out, hbuf, Wvc, bvc);
    finalize_rec  <<<BATCH, 256, 0, stream>>>(out, hbuf, cvec, btot);
    finalize_assoc<<<BATCH, 64,  0, stream>>>(out, valn);
}
